// SelfTuning_61624190763101
// MI455X (gfx1250) — matrix operand order repaired and checked by reading, not compiled, not run on hardware
//
#include <hip/hip_runtime.h>
#include <hip/hip_bf16.h>
#include <math.h>

typedef __attribute__((ext_vector_type(16))) __bf16 v16bf;
typedef __attribute__((ext_vector_type(8)))  float  v8f;

#define BATCH  256
#define DIM    1024
#define CLS    1000
#define QS     32
#define NCOL   32000           // CLS*QS
#define OUTW   32001           // 1 + CLS*QS
#define INV_T  (1.0f/0.07f)

__device__ __forceinline__ void bf16_split(float x, __bf16 &hi, __bf16 &lo) {
  hi = (__bf16)x;
  lo = (__bf16)(x - (float)hi);
}

// ---------------------------------------------------------------------------
// Kernel 1: normalize q/k, l_pos, and pack A fragments (bf16 hi/lo) in the
// CDNA5 16-bit A 16x32 WMMA lane layout:
//   lane<16 : h0..7 -> K=h,     h8..15 -> K=h+8   (K in {0..7,16..23})
//   lane>=16: h0..7 -> K=h+8,   h8..15 -> K=h+16    (K in {8..15,24..31})
// Packed as A[mtile][kstep][lane][16 halves]  (32B contiguous per lane)
// ---------------------------------------------------------------------------
__global__ __launch_bounds__(256)
void prep_kernel(const float* __restrict__ q, const float* __restrict__ k,
                 void* A_hi_v, void* A_lo_v,
                 float* __restrict__ kn, float* __restrict__ l_pos) {
  __bf16* A_hi = (__bf16*)A_hi_v;
  __bf16* A_lo = (__bf16*)A_lo_v;
  __shared__ float sQ[DIM];
  __shared__ float red[256];
  __shared__ float bq, bk;

  const int b = blockIdx.x;
  const int tid = threadIdx.x;

  float qv[4], kv[4];
  float sq = 0.f, sk = 0.f;
#pragma unroll
  for (int i = 0; i < 4; ++i) {
    int d = tid + i * 256;
    qv[i] = q[(size_t)b * DIM + d];
    kv[i] = k[(size_t)b * DIM + d];
    sq += qv[i] * qv[i];
    sk += kv[i] * kv[i];
  }
  // reduce |q|^2
  red[tid] = sq; __syncthreads();
  for (int s = 128; s > 0; s >>= 1) { if (tid < s) red[tid] += red[tid + s]; __syncthreads(); }
  if (tid == 0) bq = 1.0f / sqrtf(red[0]);
  __syncthreads();
  // reduce |k|^2
  red[tid] = sk; __syncthreads();
  for (int s = 128; s > 0; s >>= 1) { if (tid < s) red[tid] += red[tid + s]; __syncthreads(); }
  if (tid == 0) bk = 1.0f / sqrtf(red[0]);
  __syncthreads();

  const float rq = bq, rk = bk;
  float lp = 0.f;
#pragma unroll
  for (int i = 0; i < 4; ++i) {
    int d = tid + i * 256;
    float qn = qv[i] * rq;
    float knv = kv[i] * rk;
    sQ[d] = qn;
    kn[(size_t)b * DIM + d] = knv;
    lp += qn * knv;
  }
  red[tid] = lp; __syncthreads();
  for (int s = 128; s > 0; s >>= 1) { if (tid < s) red[tid] += red[tid + s]; __syncthreads(); }
  if (tid == 0) l_pos[b] = red[0];
  __syncthreads();

  // pack A fragments for this row
  const int mrow  = b & 15;
  const int mtile = b >> 4;
  for (int e = tid; e < DIM; e += 256) {
    int ks  = e >> 5;
    int idx = e & 31;
    int lane, h = idx & 15, K;
    if (idx < 16) { lane = mrow;      K = h + ((h >= 8) ? 8 : 0); }
    else          { lane = mrow + 16; K = h + 8 + ((h >= 8) ? 8 : 0); }
    float val = sQ[ks * 32 + K];
    __bf16 hi, lo; bf16_split(val, hi, lo);
    size_t off = ((((size_t)mtile * 32 + ks) * 32 + lane) * 16) + h;
    A_hi[off] = hi; A_lo[off] = lo;
  }
}

// ---------------------------------------------------------------------------
// Kernel 2: queue f32 -> packed bf16 hi/lo B fragments, via LDS transpose.
// B 32x16 layout: lane<16 -> K=h (h<8) / h+8 (h>=8) (N=lane); lane>=16 -> K=h+8 (h<8) / h+16 (h>=8) (N=lane-16).
// Packed as B[ntile][kstep][lane][16 halves]. Block tile: 32 K-rows x 64 cols.
// ---------------------------------------------------------------------------
__global__ __launch_bounds__(256)
void qpack_kernel(const float* __restrict__ queue, void* B_hi_v, void* B_lo_v) {
  __bf16* B_hi = (__bf16*)B_hi_v;
  __bf16* B_lo = (__bf16*)B_lo_v;
  __shared__ float tile[32][65];   // +1 pad vs bank conflicts

  const int nt0 = blockIdx.x * 4;  // 4 ntiles = 64 columns
  const int ks  = blockIdx.y;      // K-step (32 K rows)
  const int tid = threadIdx.x;

  for (int i = tid; i < 32 * 64; i += 256) {
    int r = i >> 6, c = i & 63;
    tile[r][c] = queue[(size_t)(ks * 32 + r) * NCOL + nt0 * 16 + c];
  }
  __syncthreads();

  for (int i = tid; i < 4 * 32 * 16; i += 256) {
    int tl   = i >> 9;
    int rem  = i & 511;
    int lane = rem >> 4;
    int h    = rem & 15;
    int K    = (h & 7) + ((h >= 8) ? 16 : 0) + ((lane >= 16) ? 8 : 0);
    int c    = tl * 16 + (lane & 15);
    float v  = tile[K][c];
    __bf16 hi, lo; bf16_split(v, hi, lo);
    size_t off = ((((size_t)(nt0 + tl)) * 32 + ks) * 32 + lane) * 16 + h;
    B_hi[off] = hi; B_lo[off] = lo;
  }
}

// ---------------------------------------------------------------------------
// Kernel 3: GEMM sims = qn @ queue via bf16x3 WMMA (hi*hi + hi*lo + lo*hi).
// grid (50, 16), 256 thr = 8 waves; wave owns 5 N-tiles of one M-tile.
// Pass-major WMMA issue keeps each acc[t] reuse 5 WMMAs apart (covers the
// BF16 WMMA->WMMA hazard window). Prefetch at WGP scope, one fragment ahead.
// ---------------------------------------------------------------------------
__global__ __launch_bounds__(256)
void gemm_kernel(const void* A_hi_v, const void* A_lo_v,
                 const void* B_hi_v, const void* B_lo_v,
                 float* __restrict__ sims) {
  const __bf16* A_hi = (const __bf16*)A_hi_v;
  const __bf16* A_lo = (const __bf16*)A_lo_v;
  const __bf16* B_hi = (const __bf16*)B_hi_v;
  const __bf16* B_lo = (const __bf16*)B_lo_v;

  const int lane   = threadIdx.x & 31;
  const int wave   = threadIdx.x >> 5;
  const int mtile  = blockIdx.y;
  const int ntbase = blockIdx.x * 40 + wave * 5;

  v8f acc[5];
#pragma unroll
  for (int t = 0; t < 5; ++t) acc[t] = (v8f){};

  for (int ks = 0; ks < 32; ++ks) {
    size_t aoff = (((size_t)mtile * 32 + ks) * 32 + lane) * 16;
    v16bf a_hi = *(const v16bf*)(A_hi + aoff);
    v16bf a_lo = *(const v16bf*)(A_lo + aoff);

    // stage all 10 B fragments for this K-step first (one load clause)
    v16bf bh[5], bl[5];
    size_t boff0 = ((((size_t)ntbase) * 32 + ks) * 32 + lane) * 16;
#pragma unroll
    for (int t = 0; t < 5; ++t) {
      size_t boff = boff0 + (size_t)t * (32 * 32 * 16);
      bh[t] = *(const v16bf*)(B_hi + boff);
      bl[t] = *(const v16bf*)(B_lo + boff);
    }

    // WGP-scope prefetch of the next K-step's first B fragment (32 lanes x
    // 32B each span the whole 1KB fragment -> whole-cacheline prefetches)
    if (ks < 31) {
      __builtin_prefetch(B_hi + boff0 + 512, 0, 3);
      __builtin_prefetch(B_lo + boff0 + 512, 0, 3);
    }

    // pass-major WMMA issue: dependency distance on each acc[t] is 5
#pragma unroll
    for (int t = 0; t < 5; ++t)
      acc[t] = __builtin_amdgcn_wmma_f32_16x16x32_bf16(false, a_hi, false, bh[t],
                                                       (short)0, acc[t], false, false);
#pragma unroll
    for (int t = 0; t < 5; ++t)
      acc[t] = __builtin_amdgcn_wmma_f32_16x16x32_bf16(false, a_hi, false, bl[t],
                                                       (short)0, acc[t], false, false);
#pragma unroll
    for (int t = 0; t < 5; ++t)
      acc[t] = __builtin_amdgcn_wmma_f32_16x16x32_bf16(false, a_lo, false, bh[t],
                                                       (short)0, acc[t], false, false);
  }

  // C/D layout: VGPR r -> M = r (lanes 0-15) / r+8 (lanes 16-31); N = lane&15
  const int m0  = mtile * 16 + ((lane >> 4) << 3);
  const int nlo = lane & 15;
#pragma unroll
  for (int t = 0; t < 5; ++t) {
    int n = (ntbase + t) * 16 + nlo;
#pragma unroll
    for (int r = 0; r < 8; ++r)
      sims[(size_t)(m0 + r) * NCOL + n] = acc[t][r];
  }
}

// ---------------------------------------------------------------------------
// Kernel 4: per-row log-softmax of [l_pos | pos(own class) | negs] / T with
// scatter into the reference ordering.
// ---------------------------------------------------------------------------
__global__ __launch_bounds__(256)
void softmax_kernel(const float* __restrict__ sims, const float* __restrict__ l_pos,
                    const int* __restrict__ labels, float* __restrict__ out_logits) {
  __shared__ float sm[256], ss[256];
  __shared__ float g_logZ;

  const int b = blockIdx.x;
  const int tid = threadIdx.x;
  const float* row = sims + (size_t)b * NCOL;

  float m = -INFINITY, s = 0.f;
  for (int j = tid; j < NCOL; j += 256) {
    float y = row[j] * INV_T;
    float M = fmaxf(m, y);
    s = s * expf(m - M) + expf(y - M);
    m = M;
  }
  sm[tid] = m; ss[tid] = s; __syncthreads();
  for (int st = 128; st > 0; st >>= 1) {
    if (tid < st) {
      float m2 = sm[tid + st], s2 = ss[tid + st];
      float M = fmaxf(sm[tid], m2);
      ss[tid] = ss[tid] * expf(sm[tid] - M) + s2 * expf(m2 - M);
      sm[tid] = M;
    }
    __syncthreads();
  }
  if (tid == 0) {
    float lp = l_pos[b] * INV_T;
    float M = fmaxf(sm[0], lp);
    float S = ss[0] * expf(sm[0] - M) + expf(lp - M);
    g_logZ = M + logf(S);
  }
  __syncthreads();

  const float logZ = g_logZ;
  const int lab = labels[b];
  float* base = out_logits + (size_t)b * OUTW;
  if (tid == 0) base[0] = l_pos[b] * INV_T - logZ;
  for (int j = tid; j < NCOL; j += 256) {
    float y = row[j] * INV_T - logZ;
    int c = j >> 5, qq = j & 31;
    int slot;
    if (c == lab) slot = 1 + qq;
    else          slot = 1 + QS + (c - (c > lab ? 1 : 0)) * QS + qq;
    base[slot] = y;
  }
}

// ---------------------------------------------------------------------------
// Kernel 5: constant pgc_labels
// ---------------------------------------------------------------------------
__global__ __launch_bounds__(256)
void labels_fill_kernel(float* __restrict__ out_labels) {
  size_t idx = (size_t)blockIdx.x * 256 + threadIdx.x;
  if (idx >= (size_t)BATCH * OUTW) return;
  int col = (int)(idx % OUTW);
  out_labels[idx] = (col < QS + 1) ? (1.0f / (QS + 1)) : 0.0f;
}

// ---------------------------------------------------------------------------
// Kernel 6: sequential enqueue semantics -> col_map + new_ptr.
// One block; thread c scans the batch in order (last-writer-wins on wrap).
// ---------------------------------------------------------------------------
__global__ __launch_bounds__(1024)
void colmap_kernel(const int* __restrict__ labels, const int* __restrict__ qptr,
                   int* __restrict__ col_map, int* __restrict__ new_ptr) {
  const int tid = threadIdx.x;
  for (int j = tid; j < NCOL; j += 1024) col_map[j] = -1;
  __syncthreads();
  if (tid < CLS) {
    int c = tid;
    int p = qptr[c];
    int cnt = 0;
    for (int i = 0; i < BATCH; ++i) {
      if (labels[i] == c) {
        col_map[c * QS + ((p + cnt) & (QS - 1))] = i;
        ++cnt;
      }
    }
    new_ptr[c] = (p + cnt) & (QS - 1);
  }
}

// ---------------------------------------------------------------------------
// Kernel 7: new_queue = queue with enqueued kn columns (fused copy+merge)
// ---------------------------------------------------------------------------
__global__ __launch_bounds__(256)
void merge_queue_kernel(const float* __restrict__ queue, const float* __restrict__ kn,
                        const int* __restrict__ col_map, float* __restrict__ out_queue) {
  size_t idx = (size_t)blockIdx.x * 256 + threadIdx.x;   // over DIM * NCOL/4
  int j4 = (int)(idx % (NCOL / 4));
  int d  = (int)(idx / (NCOL / 4));
  const float4 qv = ((const float4*)queue)[(size_t)d * (NCOL / 4) + j4];
  const int4  cm = ((const int4*)col_map)[j4];
  float4 ov;
  ov.x = (cm.x >= 0) ? kn[(size_t)cm.x * DIM + d] : qv.x;
  ov.y = (cm.y >= 0) ? kn[(size_t)cm.y * DIM + d] : qv.y;
  ov.z = (cm.z >= 0) ? kn[(size_t)cm.z * DIM + d] : qv.z;
  ov.w = (cm.w >= 0) ? kn[(size_t)cm.w * DIM + d] : qv.w;
  ((float4*)out_queue)[(size_t)d * (NCOL / 4) + j4] = ov;
}

// ---------------------------------------------------------------------------
extern "C" void kernel_launch(void* const* d_in, const int* in_sizes, int n_in,
                              void* d_out, int out_size, void* d_ws, size_t ws_size,
                              hipStream_t stream) {
  const float* q_c    = (const float*)d_in[0];
  const float* k_c    = (const float*)d_in[1];
  const float* queue  = (const float*)d_in[2];
  const int*   labels = (const int*)d_in[3];
  const int*   qptr   = (const int*)d_in[4];

  float* out          = (float*)d_out;
  float* out_logits   = out;                                   // [256][32001]
  float* out_labels   = out + (size_t)BATCH * OUTW;            // [256][32001]
  float* out_queue    = out_labels + (size_t)BATCH * OUTW;     // [1024][32000]
  int*   out_ptr      = (int*)(out_queue + (size_t)DIM * NCOL); // [1000] i32

  // workspace carve-up (all 256B aligned)
  char* w = (char*)d_ws;
  void*  A_hi  = (void*)w;  w += (size_t)BATCH * DIM * 2;        // 512 KB
  void*  A_lo  = (void*)w;  w += (size_t)BATCH * DIM * 2;        // 512 KB
  float* kn    = (float*)w; w += (size_t)BATCH * DIM * 4;        // 1 MB
  float* l_pos = (float*)w; w += 1024;
  void*  B_hi  = (void*)w;  w += (size_t)NCOL * DIM * 2;         // 64 MB
  void*  B_lo  = (void*)w;  w += (size_t)NCOL * DIM * 2;         // 64 MB
  float* sims  = (float*)w; w += (size_t)BATCH * NCOL * 4;       // 32.8 MB
  int* col_map = (int*)w;   w += (size_t)NCOL * 4;               // 128 KB

  prep_kernel<<<BATCH, 256, 0, stream>>>(q_c, k_c, A_hi, A_lo, kn, l_pos);
  qpack_kernel<<<dim3(NCOL / 64, DIM / 32), 256, 0, stream>>>(queue, B_hi, B_lo);
  gemm_kernel<<<dim3(50, 16), 256, 0, stream>>>(A_hi, A_lo, B_hi, B_lo, sims);
  softmax_kernel<<<BATCH, 256, 0, stream>>>(sims, l_pos, labels, out_logits);
  {
    size_t total = (size_t)BATCH * OUTW;
    labels_fill_kernel<<<(unsigned)((total + 255) / 256), 256, 0, stream>>>(out_labels);
  }
  colmap_kernel<<<1, 1024, 0, stream>>>(labels, qptr, col_map, out_ptr);
  merge_queue_kernel<<<(unsigned)((size_t)DIM * (NCOL / 4) / 256), 256, 0, stream>>>(
      queue, kn, col_map, out_queue);
}
